// PointNetSetAbstractionMsg_64510408786418
// MI455X (gfx1250) — compile-verified, matrix operand order checked by reading, not run on hardware
//
#include <hip/hip_runtime.h>

typedef __attribute__((ext_vector_type(16))) _Float16 v16h;
typedef __attribute__((ext_vector_type(8)))  _Float16 v8h;
typedef __attribute__((ext_vector_type(8)))  float    v8f;

constexpr int kB = 16;
constexpr int kN = 4096;
constexpr int kS = 512;
constexpr float kEPS = 1e-5f;

// ---------------------------------------------------------------------------
// Farthest point sampling: one block per batch, 512 serial argmax steps,
// running distance array kept in LDS. Also writes new_xyz (ws) and the
// (B,3,S) transposed new_xyz portion of d_out.
// ---------------------------------------------------------------------------
__global__ void fps_kernel(const float* __restrict__ xyz,   // (B,3,N)
                           int* __restrict__ fpsIdx,        // (B,S)
                           float* __restrict__ nxyz,        // (B,S,3)
                           float* __restrict__ out0)        // (B,3,S)
{
  const int b   = blockIdx.x;
  const int tid = threadIdx.x;
  __shared__ float dist[kN];
  __shared__ float rv[256];
  __shared__ int   ri[256];
  __shared__ int   sfar;

  const float* xb = xyz + (size_t)b * 3 * kN;
  for (int n = tid; n < kN; n += 256) dist[n] = 1e10f;
  if (tid == 0) sfar = 0;
  __syncthreads();

  for (int it = 0; it < kS; ++it) {
    const int far = sfar;
    if (tid == 0) fpsIdx[b * kS + it] = far;
    const float cx = xb[far], cy = xb[kN + far], cz = xb[2 * kN + far];
    float bv = -1.0f; int bi = 0;
    for (int n = tid; n < kN; n += 256) {
      const float dx = xb[n] - cx, dy = xb[kN + n] - cy, dz = xb[2 * kN + n] - cz;
      const float d  = dx * dx + dy * dy + dz * dz;
      const float dm = fminf(dist[n], d);
      dist[n] = dm;
      if (dm > bv) { bv = dm; bi = n; }
    }
    rv[tid] = bv; ri[tid] = bi;
    __syncthreads();
    for (int st = 128; st > 0; st >>= 1) {
      if (tid < st) {
        const float v2 = rv[tid + st]; const int i2 = ri[tid + st];
        if (v2 > rv[tid] || (v2 == rv[tid] && i2 < ri[tid])) { rv[tid] = v2; ri[tid] = i2; }
      }
      __syncthreads();
    }
    if (tid == 0) sfar = ri[0];
    __syncthreads();
  }

  for (int s = tid; s < kS; s += 256) {
    const int idx = fpsIdx[b * kS + s];
    const float x = xb[idx], y = xb[kN + idx], z = xb[2 * kN + idx];
    float* nb = nxyz + ((size_t)b * kS + s) * 3;
    nb[0] = x; nb[1] = y; nb[2] = z;
    float* ob = out0 + (size_t)b * 3 * kS;
    ob[0 * kS + s] = x; ob[1 * kS + s] = y; ob[2 * kS + s] = z;
  }
}

// ---------------------------------------------------------------------------
// Ball query, all 3 radii in one scan (r0^2 < r1^2 < r2^2 are nested).
// One thread per (b,s); batch xyz cached in LDS. Indices naturally ascending
// (matches jnp.sort-of-masked-indices semantics), padded with first hit.
// ---------------------------------------------------------------------------
__global__ void ballquery_kernel(const float* __restrict__ xyz,
                                 const float* __restrict__ nxyz,
                                 int* __restrict__ g0, int* __restrict__ g1,
                                 int* __restrict__ g2)
{
  const int b = blockIdx.x >> 1;
  const int s = ((blockIdx.x & 1) << 8) + threadIdx.x;
  __shared__ float sx[kN], sy[kN], sz[kN];
  const float* xb = xyz + (size_t)b * 3 * kN;
  for (int n = threadIdx.x; n < kN; n += 256) {
    sx[n] = xb[n]; sy[n] = xb[kN + n]; sz[n] = xb[2 * kN + n];
  }
  __syncthreads();

  const size_t bs = (size_t)b * kS + s;
  const float cx = nxyz[bs * 3 + 0], cy = nxyz[bs * 3 + 1], cz = nxyz[bs * 3 + 2];
  int* p0 = g0 + bs * 16;
  int* p1 = g1 + bs * 32;
  int* p2 = g2 + bs * 128;
  int c0 = 0, c1 = 0, c2 = 0;
  for (int n = 0; n < kN; ++n) {
    const float dx = sx[n] - cx, dy = sy[n] - cy, dz = sz[n] - cz;
    const float d2 = dx * dx + dy * dy + dz * dz;
    if (d2 <= 0.16f) {
      if (c2 < 128) p2[c2++] = n;
      if (d2 <= 0.04f) {
        if (c1 < 32) p1[c1++] = n;
        if (d2 <= 0.01f && c0 < 16) p0[c0++] = n;
      }
    }
  }
  const int f0 = c0 ? p0[0] : 0;
  const int f1 = c1 ? p1[0] : 0;
  const int f2 = c2 ? p2[0] : 0;
  for (int k = c0; k < 16;  ++k) p0[k] = f0;
  for (int k = c1; k < 32;  ++k) p1[k] = f1;
  for (int k = c2; k < 128; ++k) p2[k] = f2;
}

// ---------------------------------------------------------------------------
// Gather + relative-xyz concat: act0[row][0:3] = point feats, [3:6] = xyz-rel,
// [6:32] = 0 (K-pad for the first 16x16x32 WMMA layer). f16 output.
// ---------------------------------------------------------------------------
__global__ void group_kernel(const float* __restrict__ xyz,
                             const float* __restrict__ pts,
                             const float* __restrict__ nxyz,
                             const int* __restrict__ gidx,
                             int Kk, _Float16* __restrict__ act)
{
  const int t = blockIdx.x * blockDim.x + threadIdx.x;
  const int total = kB * kS * Kk;
  if (t >= total) return;
  const int bs = t / Kk;
  const int b  = bs / kS;
  int n = gidx[t];
  n = n < 0 ? 0 : (n > kN - 1 ? kN - 1 : n);
  const float* xb = xyz + (size_t)b * 3 * kN;
  const float* pb = pts + (size_t)b * 3 * kN;
  _Float16* dst = act + (size_t)t * 32;
  dst[0] = (_Float16)pb[n];
  dst[1] = (_Float16)pb[kN + n];
  dst[2] = (_Float16)pb[2 * kN + n];
  dst[3] = (_Float16)(xb[n]          - nxyz[(size_t)bs * 3 + 0]);
  dst[4] = (_Float16)(xb[kN + n]     - nxyz[(size_t)bs * 3 + 1]);
  dst[5] = (_Float16)(xb[2 * kN + n] - nxyz[(size_t)bs * 3 + 2]);
#pragma unroll
  for (int c = 6; c < 32; ++c) dst[c] = (_Float16)0.f;
}

// ---------------------------------------------------------------------------
// Pre-swizzle W (Cout x CinReal, row major f32) into per-lane WMMA B tiles:
// tile (nt,kt) is 512 halves; lane L holds column o = nt*16 + L%16, with
// K halves [kbase..kbase+7, kbase+16..kbase+23], kbase = (L/16)*8
// (mirrors the documented 16-bit A layout). Zero-padded outside real dims.
// ---------------------------------------------------------------------------
__global__ void pack_w_kernel(const float* __restrict__ W, int cinReal, int cout,
                              _Float16* __restrict__ packW, int nKt)
{
  const int tile = blockIdx.x;           // nt*nKt + kt
  const int lane = threadIdx.x;          // 0..31
  const int nt = tile / nKt, kt = tile % nKt;
  const int o = nt * 16 + (lane & 15);
  const int kbase = (lane >> 4) * 8;
  _Float16* dst = packW + (size_t)tile * 512 + lane * 16;
#pragma unroll
  for (int e = 0; e < 16; ++e) {
    const int kk = (e < 8) ? (kbase + e) : (kbase + 16 + (e - 8));
    const int c  = kt * 32 + kk;
    const float v = (c < cinReal && o < cout) ? W[o * cinReal + c] : 0.f;
    dst[e] = (_Float16)v;
  }
}

__global__ void zero_kernel(float* p, int n)
{
  const int i = blockIdx.x * blockDim.x + threadIdx.x;
  if (i < n) p[i] = 0.f;
}

__global__ void bn_finalize_kernel(const float* __restrict__ stats,
                                   const float* __restrict__ gamma,
                                   const float* __restrict__ beta,
                                   float invM, int cout, float* __restrict__ scsh)
{
  const int o = blockIdx.x * blockDim.x + threadIdx.x;
  if (o >= cout) return;
  const float mu  = stats[o] * invM;
  const float var = stats[cout + o] * invM - mu * mu;
  const float sc  = gamma[o] * rsqrtf(var + kEPS);
  scsh[o]        = sc;
  scsh[cout + o] = beta[o] - mu * sc;
}

// ---------------------------------------------------------------------------
// WMMA GEMM, fully register-resident: NKT = cinPad/32 is a template constant,
// so each wave front-loads ALL B tiles (NT x NKT x v16h, broadcast weights)
// and ALL A chunks (NKT x v16h) in one load clause, then runs an
// uninterrupted chain of NT*NKT v_wmma_f32_16x16x32_f16 ops with no memory
// waits or branches in between. Wave = 16(M) x NT*16(N) output strip.
// All 8 waves of a block share one N-group: MODE 0 reduces per-channel
// sum/sumsq in LDS (ds_add_f32) then issues 2*NT*16 global atomics/block.
// MODE 1: y = relu(h*scale + shift) -> f16 activations for the next layer.
// tilesM = mrows/16 is always a multiple of 8 here, so the mapping is exact
// and wave-uniform (EXEC all-ones around WMMA, no divergent barriers).
// ---------------------------------------------------------------------------
template <int MODE, int NT, int NKT>
__global__ void gemm_wmma_kernel(const _Float16* __restrict__ actIn,
                                 const _Float16* __restrict__ packW,
                                 int nGroups,
                                 const float* __restrict__ bias, int cout,
                                 float* __restrict__ stats,
                                 const float* __restrict__ scsh,
                                 _Float16* __restrict__ actOut)
{
  const int waveId = threadIdx.x >> 5;                 // 0..7
  const int lane   = threadIdx.x & 31;
  const int ng     = blockIdx.x % nGroups;             // N-group (block-uniform)
  const int mt     = (blockIdx.x / nGroups) * 8 + waveId;

  const int hi   = lane >> 4;                          // K-half select
  const int mrow = (mt << 4) + (lane & 15);
  const _Float16* aBase = actIn + (size_t)mrow * (NKT * 32) + hi * 8;
  const int ntBase = ng * NT;
  const _Float16* bBase = packW + (size_t)ntBase * NKT * 512 + lane * 16;

  // Front-load all operands (single clause, one wait).
  v16h bt[NT][NKT];
#pragma unroll
  for (int j = 0; j < NT; ++j)
#pragma unroll
    for (int kt = 0; kt < NKT; ++kt)
      bt[j][kt] = *(const v16h*)(bBase + ((size_t)j * NKT + kt) * 512);

  v16h at[NKT];
#pragma unroll
  for (int kt = 0; kt < NKT; ++kt) {
    const v8h lo = *(const v8h*)(aBase + kt * 32);
    const v8h hh = *(const v8h*)(aBase + kt * 32 + 16);
    at[kt] = __builtin_shufflevector(lo, hh, 0, 1, 2, 3, 4, 5, 6, 7,
                                             8, 9, 10, 11, 12, 13, 14, 15);
  }

  v8f acc[NT];
#pragma unroll
  for (int j = 0; j < NT; ++j) { v8f z = {}; acc[j] = z; }

  // Back-to-back matrix-core chain, no intervening memory ops.
#pragma unroll
  for (int kt = 0; kt < NKT; ++kt)
#pragma unroll
    for (int j = 0; j < NT; ++j)
      acc[j] = __builtin_amdgcn_wmma_f32_16x16x32_f16(false, at[kt], false, bt[j][kt],
                                                      (short)0, acc[j], false, false);

  if constexpr (MODE == 0) {
    __shared__ float ssum[NT * 16];
    __shared__ float ssq [NT * 16];
    for (int i = threadIdx.x; i < NT * 16; i += blockDim.x) { ssum[i] = 0.f; ssq[i] = 0.f; }
    __syncthreads();
#pragma unroll
    for (int j = 0; j < NT; ++j) {
      const float bo = bias[(ntBase + j) * 16 + (lane & 15)];
      float s1 = 0.f, s2 = 0.f;
#pragma unroll
      for (int i = 0; i < 8; ++i) { const float h = acc[j][i] + bo; s1 += h; s2 += h * h; }
      s1 += __shfl_xor(s1, 16);                        // lanes L and L^16 share col
      s2 += __shfl_xor(s2, 16);
      if (hi == 0) {
        atomicAdd(&ssum[j * 16 + (lane & 15)], s1);    // ds_add_f32
        atomicAdd(&ssq [j * 16 + (lane & 15)], s2);
      }
    }
    __syncthreads();
    if (threadIdx.x < NT * 16) {
      const int col = ntBase * 16 + threadIdx.x;
      atomicAdd(&stats[col],        ssum[threadIdx.x]);
      atomicAdd(&stats[cout + col], ssq [threadIdx.x]);
    }
  } else {
#pragma unroll
    for (int j = 0; j < NT; ++j) {
      const int col = (ntBase + j) * 16 + (lane & 15);
      const float bo = bias[col];
      const float sc = scsh[col], sh = scsh[cout + col];
#pragma unroll
      for (int i = 0; i < 8; ++i) {
        float h = (acc[j][i] + bo) * sc + sh;
        h = fmaxf(h, 0.f);
        const int r = (mt << 4) + hi * 8 + i;
        actOut[(size_t)r * cout + col] = (_Float16)h;
      }
    }
  }
}

// Dispatch over the (NKT, NT) combos actually used by the 9 layers.
template <int MODE>
static void launch_gemm(int NT, int nKt, int blocks, hipStream_t stream,
                        const _Float16* actIn, const _Float16* packW, int nGroups,
                        const float* bias, int cout,
                        float* stats, const float* scsh, _Float16* actOut)
{
  switch (nKt * 10 + NT) {
    case 12: gemm_wmma_kernel<MODE, 2, 1><<<blocks, 256, 0, stream>>>(
                 actIn, packW, nGroups, bias, cout, stats, scsh, actOut); break;
    case 13: gemm_wmma_kernel<MODE, 3, 1><<<blocks, 256, 0, stream>>>(
                 actIn, packW, nGroups, bias, cout, stats, scsh, actOut); break;
    case 14: gemm_wmma_kernel<MODE, 4, 1><<<blocks, 256, 0, stream>>>(
                 actIn, packW, nGroups, bias, cout, stats, scsh, actOut); break;
    case 22: gemm_wmma_kernel<MODE, 2, 2><<<blocks, 256, 0, stream>>>(
                 actIn, packW, nGroups, bias, cout, stats, scsh, actOut); break;
    case 23: gemm_wmma_kernel<MODE, 3, 2><<<blocks, 256, 0, stream>>>(
                 actIn, packW, nGroups, bias, cout, stats, scsh, actOut); break;
    case 24: gemm_wmma_kernel<MODE, 4, 2><<<blocks, 256, 0, stream>>>(
                 actIn, packW, nGroups, bias, cout, stats, scsh, actOut); break;
    case 33: gemm_wmma_kernel<MODE, 3, 3><<<blocks, 256, 0, stream>>>(
                 actIn, packW, nGroups, bias, cout, stats, scsh, actOut); break;
    case 34: gemm_wmma_kernel<MODE, 4, 3><<<blocks, 256, 0, stream>>>(
                 actIn, packW, nGroups, bias, cout, stats, scsh, actOut); break;
    default: break;  // unreachable for this network
  }
}

// ---------------------------------------------------------------------------
// Max over the K neighbors -> d_out new_points slice (B,320,S).
// ---------------------------------------------------------------------------
__global__ void maxpool_kernel(const _Float16* __restrict__ act, int Kk, int cout,
                               float* __restrict__ outPts, int chBase)
{
  const int t = blockIdx.x * blockDim.x + threadIdx.x;
  const int total = kB * kS * cout;
  if (t >= total) return;
  const int o  = t % cout;
  const int bs = t / cout;
  const int b = bs / kS, s = bs % kS;
  const _Float16* ab = act + (size_t)bs * Kk * cout + o;
  float m = -3.4e38f;
  for (int k = 0; k < Kk; ++k) m = fmaxf(m, (float)ab[(size_t)k * cout]);
  outPts[(size_t)b * 320 * kS + (size_t)(chBase + o) * kS + s] = m;
}

// ---------------------------------------------------------------------------
extern "C" void kernel_launch(void* const* d_in, const int* in_sizes, int n_in,
                              void* d_out, int out_size, void* d_ws, size_t ws_size,
                              hipStream_t stream)
{
  (void)in_sizes; (void)n_in; (void)out_size; (void)ws_size;
  const float* xyz = (const float*)d_in[0];
  const float* pts = (const float*)d_in[1];
  auto P = [&](int br, int ly, int t) -> const float* {
    return (const float*)d_in[2 + (br * 3 + ly) * 4 + t];   // W,b,gamma,beta
  };

  float* out    = (float*)d_out;
  float* outPts = out + (size_t)kB * 3 * kS;

  char* base = (char*)d_ws;
  size_t off = 0;
  auto alloc = [&](size_t bytes) -> void* {
    void* p = base + off;
    off = (off + bytes + 255) & ~(size_t)255;
    return p;
  };
  int*      fpsIdx = (int*)   alloc((size_t)kB * kS * 4);
  float*    nxyz   = (float*) alloc((size_t)kB * kS * 3 * 4);
  int*      g0     = (int*)   alloc((size_t)kB * kS * 16 * 4);
  int*      g1     = (int*)   alloc((size_t)kB * kS * 32 * 4);
  int*      g2     = (int*)   alloc((size_t)kB * kS * 128 * 4);
  float*    stats  = (float*) alloc(256 * 4);
  float*    scsh   = (float*) alloc(256 * 4);
  _Float16* packW  = (_Float16*)alloc((size_t)32768 * 2);
  const size_t Mmax = (size_t)kB * kS * 128;
  _Float16* actA = (_Float16*)alloc(Mmax * 96 * 2);    // layer-1 outputs <= 96ch
  _Float16* actB = (_Float16*)alloc(Mmax * 128 * 2);   // layer-0/2 outputs <= 128ch

  fps_kernel<<<kB, 256, 0, stream>>>(xyz, fpsIdx, nxyz, out);
  ballquery_kernel<<<kB * 2, 256, 0, stream>>>(xyz, nxyz, g0, g1, g2);

  const int KS[3]     = {16, 32, 128};
  const int MLP[3][3] = {{32, 32, 64}, {64, 64, 128}, {64, 96, 128}};
  int* gArr[3] = {g0, g1, g2};
  int chBase = 0;

  for (int br = 0; br < 3; ++br) {
    const int Kk = KS[br];
    const int mrows = kB * kS * Kk;

    group_kernel<<<(mrows + 255) / 256, 256, 0, stream>>>(xyz, pts, nxyz,
                                                          gArr[br], Kk, actA);
    _Float16* cur = actA;
    _Float16* nxt = actB;
    int cinReal = 6, cinPad = 32;
    for (int ly = 0; ly < 3; ++ly) {
      const int cout = MLP[br][ly];
      const int nKt = cinPad / 32, nNt = cout / 16;
      pack_w_kernel<<<nNt * nKt, 32, 0, stream>>>(P(br, ly, 0), cinReal, cout,
                                                  packW, nKt);
      zero_kernel<<<1, 256, 0, stream>>>(stats, 2 * cout);

      const int NT      = (cout == 32) ? 2 : (cout == 96) ? 3 : 4;
      const int nGroups = nNt / NT;
      const int blocks  = nGroups * (mrows / 16 / 8);   // tilesM % 8 == 0 always
      const float* Wb = P(br, ly, 1);

      launch_gemm<0>(NT, nKt, blocks, stream, cur, packW, nGroups, Wb, cout,
                     stats, nullptr, nullptr);
      bn_finalize_kernel<<<1, 128, 0, stream>>>(stats, P(br, ly, 2), P(br, ly, 3),
                                                1.0f / (float)mrows, cout, scsh);
      launch_gemm<1>(NT, nKt, blocks, stream, cur, packW, nGroups, Wb, cout,
                     nullptr, scsh, nxt);

      _Float16* tswap = cur; cur = nxt; nxt = tswap;
      cinReal = cout; cinPad = cout;
    }
    const int coutF = MLP[br][2];
    maxpool_kernel<<<(kB * kS * coutF + 255) / 256, 256, 0, stream>>>(cur, Kk, coutF,
                                                                      outPts, chBase);
    chBase += coutF;
  }
}
